// LayerNormMoELayer_45294725104293
// MI455X (gfx1250) — matrix operand order repaired and checked by reading, not compiled, not run on hardware
//
#include <hip/hip_runtime.h>

#define HIDDEN 1024
#define FFN    2816
#define NEXP   8
#define T_TOK  4096

#define TM 128
#define TN 64
#define KS 32

typedef __attribute__((ext_vector_type(16))) __bf16          v16bf;
typedef __attribute__((ext_vector_type(16))) unsigned short  v16u;
typedef __attribute__((ext_vector_type(8)))  unsigned short  v8u;
typedef __attribute__((ext_vector_type(8)))  float           v8f;

typedef __attribute__((address_space(3))) unsigned short lds_ushort;

union V16 { v16u u; v16bf b; };

// pack two floats into one dword of bf16 (RNE via compiler's native cvt)
__device__ __forceinline__ unsigned pk_bf16(float lo, float hi) {
  union { __bf16 h[2]; unsigned u; } p;
  p.h[0] = (__bf16)lo; p.h[1] = (__bf16)hi;
  return p.u;
}
__device__ __forceinline__ unsigned short bf16bits(float f) {
  union { __bf16 h; unsigned short u; } p;
  p.h = (__bf16)f; return p.u;
}

// async copy 16B global -> LDS (ASYNCcnt-tracked)
__device__ __forceinline__ void async_g2l_b128(lds_ushort* l, const unsigned short* g) {
  asm volatile("global_load_async_to_lds_b128 %0, %1, off"
               :: "v"(l), "v"(g) : "memory");
}
__device__ __forceinline__ void wait_async0() {
  asm volatile("s_wait_asynccnt 0x0" ::: "memory");
}

// ---------------------------------------------------------------- init
__global__ __launch_bounds__(256) void moe_init_kernel(float* __restrict__ out,
                                                       int* __restrict__ counts) {
  size_t idx = (size_t)blockIdx.x * 1024 + (size_t)threadIdx.x * 4;
  float4 z = {0.f, 0.f, 0.f, 0.f};
  *(float4*)(out + idx) = z;
  if (blockIdx.x == 0 && threadIdx.x < NEXP) counts[threadIdx.x] = 0;
}

// ---------------------------------------------------------------- router + RMSNorm
__global__ __launch_bounds__(256) void moe_router_kernel(
    const float* __restrict__ hs, const float* __restrict__ lnw,
    const float* __restrict__ rw, unsigned short* __restrict__ xb,
    int* __restrict__ counts, int* __restrict__ token_list,
    float* __restrict__ gate_list)
{
  const int t    = blockIdx.x;
  const int tid  = threadIdx.x;
  const int lane = tid & 31, wid = tid >> 5;

  __shared__ float xn[HIDDEN];
  __shared__ float red[8];
  __shared__ float lg[NEXP];

  const float4 v = ((const float4*)(hs + (size_t)t * HIDDEN))[tid];
  float ss = v.x*v.x + v.y*v.y + v.z*v.z + v.w*v.w;
  #pragma unroll
  for (int o = 16; o > 0; o >>= 1) ss += __shfl_xor(ss, o);
  if (lane == 0) red[wid] = ss;
  __syncthreads();
  if (tid == 0) {
    float s = 0.f;
    #pragma unroll
    for (int i = 0; i < 8; ++i) s += red[i];
    red[0] = rsqrtf(s / (float)HIDDEN + 1e-5f);
  }
  __syncthreads();
  const float rs = red[0];

  const float4 w4 = ((const float4*)lnw)[tid];
  float4 nv;
  nv.x = v.x * rs * w4.x;  nv.y = v.y * rs * w4.y;
  nv.z = v.z * rs * w4.z;  nv.w = v.w * rs * w4.w;
  ((float4*)xn)[tid] = nv;

  unsigned* dst = (unsigned*)(xb + (size_t)t * HIDDEN + (size_t)tid * 4);
  dst[0] = pk_bf16(nv.x, nv.y);
  dst[1] = pk_bf16(nv.z, nv.w);
  __syncthreads();

  // wave `wid` computes router logit `wid`
  float acc = 0.f;
  for (int j = lane; j < HIDDEN; j += 32) acc += xn[j] * rw[j * NEXP + wid];
  #pragma unroll
  for (int o = 16; o > 0; o >>= 1) acc += __shfl_xor(acc, o);
  if (lane == 0) lg[wid] = acc;
  __syncthreads();

  if (tid == 0) {
    float mx = lg[0];
    #pragma unroll
    for (int i = 1; i < NEXP; ++i) mx = fmaxf(mx, lg[i]);
    float p[NEXP]; float s = 0.f;
    #pragma unroll
    for (int i = 0; i < NEXP; ++i) { p[i] = __expf(lg[i] - mx); s += p[i]; }
    const float inv = 1.f / s;
    int i1 = 0; float b1 = p[0];
    #pragma unroll
    for (int i = 1; i < NEXP; ++i) if (p[i] > b1) { b1 = p[i]; i1 = i; }
    int i2 = (i1 == 0) ? 1 : 0; float b2 = p[i2];
    #pragma unroll
    for (int i = 0; i < NEXP; ++i) if (i != i1 && p[i] > b2) { b2 = p[i]; i2 = i; }
    const int   es[2] = { i1, i2 };
    const float gs[2] = { b1 * inv, b2 * inv };
    #pragma unroll
    for (int k = 0; k < 2; ++k) {
      int pos = atomicAdd(&counts[es[k]], 1);
      token_list[es[k] * T_TOK + pos] = t;
      gate_list[es[k] * T_TOK + pos]  = gs[k];
    }
  }
}

// ---------------------------------------------------------------- offsets (scan of 8)
__global__ void moe_offsets_kernel(const int* __restrict__ counts,
                                   int* __restrict__ offsets) {
  if (threadIdx.x == 0) {
    int s = 0;
    for (int e = 0; e < NEXP; ++e) { offsets[e] = s; s += counts[e]; }
  }
}

// ---------------------------------------------------------------- FFN stage 1: act = silu(x@w1)*(x@w3)
__global__ __launch_bounds__(256) void moe_ffn1_kernel(
    const unsigned short* __restrict__ xb,
    const float* __restrict__ w1, const float* __restrict__ w3,
    const int* __restrict__ counts, const int* __restrict__ offsets,
    const int* __restrict__ token_list,
    unsigned short* __restrict__ act)
{
  const int e   = blockIdx.z;
  const int cnt = counts[e];
  const int mt  = blockIdx.y;
  if (mt * TM >= cnt) return;
  const int nt    = blockIdx.x;
  const int tid   = threadIdx.x;
  const int lane  = tid & 31, wid = tid >> 5;
  const int wm    = wid & 3,  wn  = wid >> 2;
  const int lrow  = lane & 15, lhalf = lane >> 4;

  __shared__ unsigned short lA [TM][KS];
  __shared__ unsigned short lB1[TN][KS];   // transposed [n][k]
  __shared__ unsigned short lB3[TN][KS];
  __shared__ int toks[TM];

  if (tid < TM) {
    int row = mt * TM + tid;
    toks[tid] = (row < cnt) ? token_list[e * T_TOK + row] : -1;
  }
  __syncthreads();

  v8f z = {0.f,0.f,0.f,0.f,0.f,0.f,0.f,0.f};
  v8f acc1[2][2], acc3[2][2];
  #pragma unroll
  for (int mi = 0; mi < 2; ++mi)
    #pragma unroll
    for (int ni = 0; ni < 2; ++ni) { acc1[mi][ni] = z; acc3[mi][ni] = z; }

  const float* w1e = w1 + (size_t)e * HIDDEN * FFN;
  const float* w3e = w3 + (size_t)e * HIDDEN * FFN;

  // per-thread staging coordinates
  const int arow = tid >> 1, ahalf = tid & 1;      // A: 128 rows x 2 halves of 16 bf16
  const int bk   = (tid >> 4) * 2;                 // B: k-pair 0..30
  const int bn   = (tid & 15) * 4;                 // B: n-quad

  for (int k0 = 0; k0 < HIDDEN; k0 += KS) {
    { // stage A via async global->LDS (gathered token rows, already bf16)
      int tok = toks[arow];
      lds_ushort* l0 = (lds_ushort*)&lA[arow][ahalf * 16];
      if (tok >= 0) {
        const unsigned short* g = xb + (size_t)tok * HIDDEN + k0 + ahalf * 16;
        async_g2l_b128(l0,     g);
        async_g2l_b128(l0 + 8, g + 8);
      } else {
        uint4 zz = {0,0,0,0};
        *(uint4*)&lA[arow][ahalf * 16]     = zz;
        *(uint4*)&lA[arow][ahalf * 16 + 8] = zz;
      }
    }
    { // stage B: f32 -> packed bf16 pairs, transposed to [n][k]
      size_t g0 = (size_t)(k0 + bk) * FFN + (size_t)nt * TN + bn;
      float4 a1 = *(const float4*)(w1e + g0);
      float4 c1 = *(const float4*)(w1e + g0 + FFN);
      float4 a3 = *(const float4*)(w3e + g0);
      float4 c3 = *(const float4*)(w3e + g0 + FFN);
      if (k0 + KS < HIDDEN) {                 // prefetch next K-step tile
        __builtin_prefetch(w1e + g0 + (size_t)KS * FFN, 0, 1);
        __builtin_prefetch(w3e + g0 + (size_t)KS * FFN, 0, 1);
      }
      *(unsigned*)&lB1[bn+0][bk] = pk_bf16(a1.x, c1.x);
      *(unsigned*)&lB1[bn+1][bk] = pk_bf16(a1.y, c1.y);
      *(unsigned*)&lB1[bn+2][bk] = pk_bf16(a1.z, c1.z);
      *(unsigned*)&lB1[bn+3][bk] = pk_bf16(a1.w, c1.w);
      *(unsigned*)&lB3[bn+0][bk] = pk_bf16(a3.x, c3.x);
      *(unsigned*)&lB3[bn+1][bk] = pk_bf16(a3.y, c3.y);
      *(unsigned*)&lB3[bn+2][bk] = pk_bf16(a3.z, c3.z);
      *(unsigned*)&lB3[bn+3][bk] = pk_bf16(a3.w, c3.w);
    }
    wait_async0();
    __syncthreads();

    V16 af[2];
    #pragma unroll
    for (int mi = 0; mi < 2; ++mi) {
      const int m = wm * 32 + mi * 16 + lrow;
      v8u lo = *(const v8u*)&lA[m][lhalf * 16];
      v8u hi = *(const v8u*)&lA[m][lhalf * 16 + 8];
      #pragma unroll
      for (int i = 0; i < 8; ++i) { af[mi].u[i] = lo[i]; af[mi].u[i + 8] = hi[i]; }
    }
    V16 bfr1[2], bfr3[2];
    #pragma unroll
    for (int ni = 0; ni < 2; ++ni) {
      const int n = wn * 32 + ni * 16 + lrow;
      bfr1[ni].u = *(const v16u*)&lB1[n][lhalf * 16];
      bfr3[ni].u = *(const v16u*)&lB3[n][lhalf * 16];
    }
    #pragma unroll
    for (int mi = 0; mi < 2; ++mi)
      #pragma unroll
      for (int ni = 0; ni < 2; ++ni) {
        acc1[mi][ni] = __builtin_amdgcn_wmma_f32_16x16x32_bf16(
            false, af[mi].b, false, bfr1[ni].b, (short)0, acc1[mi][ni], false, false);
        acc3[mi][ni] = __builtin_amdgcn_wmma_f32_16x16x32_bf16(
            false, af[mi].b, false, bfr3[ni].b, (short)0, acc3[mi][ni], false, false);
      }
    __syncthreads();
  }

  const size_t abase = (size_t)offsets[e];
  #pragma unroll
  for (int mi = 0; mi < 2; ++mi) {
    #pragma unroll
    for (int r = 0; r < 8; ++r) {
      int mlocal = mt * TM + wm * 32 + mi * 16 + r + lhalf * 8;
      if (mlocal < cnt) {
        #pragma unroll
        for (int ni = 0; ni < 2; ++ni) {
          int col = nt * TN + wn * 32 + ni * 16 + lrow;
          float h1v = acc1[mi][ni][r];
          float h3v = acc3[mi][ni][r];
          float sv  = h1v / (1.f + __expf(-h1v)) * h3v;   // silu(h1)*h3
          act[(abase + (size_t)mlocal) * FFN + col] = bf16bits(sv);
        }
      }
    }
  }
}

// ---------------------------------------------------------------- FFN stage 2: out += gate * (act @ w2)
__global__ __launch_bounds__(256) void moe_ffn2_kernel(
    const unsigned short* __restrict__ act,
    const float* __restrict__ w2,
    const int* __restrict__ counts, const int* __restrict__ offsets,
    const int* __restrict__ token_list, const float* __restrict__ gate_list,
    float* __restrict__ out)
{
  const int e   = blockIdx.z;
  const int cnt = counts[e];
  const int mt  = blockIdx.y;
  if (mt * TM >= cnt) return;
  const int nt    = blockIdx.x;
  const int tid   = threadIdx.x;
  const int lane  = tid & 31, wid = tid >> 5;
  const int wm    = wid & 3,  wn  = wid >> 2;
  const int lrow  = lane & 15, lhalf = lane >> 4;

  __shared__ unsigned short lA[TM][KS];
  __shared__ unsigned short lB[TN][KS];    // transposed [n][k]

  const size_t abase = (size_t)offsets[e];
  const float* w2e = w2 + (size_t)e * FFN * HIDDEN;

  v8f z = {0.f,0.f,0.f,0.f,0.f,0.f,0.f,0.f};
  v8f acc[2][2];
  #pragma unroll
  for (int mi = 0; mi < 2; ++mi)
    #pragma unroll
    for (int ni = 0; ni < 2; ++ni) acc[mi][ni] = z;

  const int arow = tid >> 1, ahalf = tid & 1;
  const int bk   = (tid >> 4) * 2;
  const int bn   = (tid & 15) * 4;
  const int mrow = mt * TM + arow;

  for (int k0 = 0; k0 < FFN; k0 += KS) {   // 88 steps
    { // stage A via async global->LDS (act rows, bf16)
      lds_ushort* l0 = (lds_ushort*)&lA[arow][ahalf * 16];
      if (mrow < cnt) {
        const unsigned short* g = act + (abase + (size_t)mrow) * FFN + k0 + ahalf * 16;
        async_g2l_b128(l0,     g);
        async_g2l_b128(l0 + 8, g + 8);
      } else {
        uint4 zz = {0,0,0,0};
        *(uint4*)&lA[arow][ahalf * 16]     = zz;
        *(uint4*)&lA[arow][ahalf * 16 + 8] = zz;
      }
    }
    { // stage B: w2 f32 -> packed bf16 pairs, transposed
      size_t g0 = (size_t)(k0 + bk) * HIDDEN + (size_t)nt * TN + bn;
      float4 a2 = *(const float4*)(w2e + g0);
      float4 c2 = *(const float4*)(w2e + g0 + HIDDEN);
      if (k0 + KS < FFN)
        __builtin_prefetch(w2e + g0 + (size_t)KS * HIDDEN, 0, 1);
      *(unsigned*)&lB[bn+0][bk] = pk_bf16(a2.x, c2.x);
      *(unsigned*)&lB[bn+1][bk] = pk_bf16(a2.y, c2.y);
      *(unsigned*)&lB[bn+2][bk] = pk_bf16(a2.z, c2.z);
      *(unsigned*)&lB[bn+3][bk] = pk_bf16(a2.w, c2.w);
    }
    wait_async0();
    __syncthreads();

    V16 af[2];
    #pragma unroll
    for (int mi = 0; mi < 2; ++mi) {
      const int m = wm * 32 + mi * 16 + lrow;
      v8u lo = *(const v8u*)&lA[m][lhalf * 16];
      v8u hi = *(const v8u*)&lA[m][lhalf * 16 + 8];
      #pragma unroll
      for (int i = 0; i < 8; ++i) { af[mi].u[i] = lo[i]; af[mi].u[i + 8] = hi[i]; }
    }
    V16 bfr[2];
    #pragma unroll
    for (int ni = 0; ni < 2; ++ni) {
      const int n = wn * 32 + ni * 16 + lrow;
      bfr[ni].u = *(const v16u*)&lB[n][lhalf * 16];
    }
    #pragma unroll
    for (int mi = 0; mi < 2; ++mi)
      #pragma unroll
      for (int ni = 0; ni < 2; ++ni)
        acc[mi][ni] = __builtin_amdgcn_wmma_f32_16x16x32_bf16(
            false, af[mi].b, false, bfr[ni].b, (short)0, acc[mi][ni], false, false);
    __syncthreads();
  }

  #pragma unroll
  for (int mi = 0; mi < 2; ++mi) {
    #pragma unroll
    for (int r = 0; r < 8; ++r) {
      int mlocal = mt * TM + wm * 32 + mi * 16 + r + lhalf * 8;
      if (mlocal < cnt) {
        int   tok = token_list[e * T_TOK + mlocal];
        float g   = gate_list[e * T_TOK + mlocal];
        #pragma unroll
        for (int ni = 0; ni < 2; ++ni) {
          int col = nt * TN + wn * 32 + ni * 16 + lrow;
          atomicAdd(&out[(size_t)tok * HIDDEN + col], g * acc[mi][ni][r]);
        }
      }
    }
  }
}

// ---------------------------------------------------------------- launch
extern "C" void kernel_launch(void* const* d_in, const int* in_sizes, int n_in,
                              void* d_out, int out_size, void* d_ws, size_t ws_size,
                              hipStream_t stream) {
  const float* hs  = (const float*)d_in[0];   // [2,2048,1024]
  const float* lnw = (const float*)d_in[1];   // [1024]
  const float* rw  = (const float*)d_in[2];   // [1024,8]
  const float* w1  = (const float*)d_in[3];   // [8,1024,2816]
  const float* w3  = (const float*)d_in[4];   // [8,1024,2816]
  const float* w2  = (const float*)d_in[5];   // [8,2816,1024]
  float* out = (float*)d_out;                 // [2,2048,1024]

  char* ws = (char*)d_ws;
  int*   counts     = (int*)(ws);                                    // 32 B
  int*   offsets    = (int*)(ws + 128);                              // 32 B
  int*   token_list = (int*)(ws + 256);                              // 8*4096*4
  float* gate_list  = (float*)(ws + 256 + (size_t)NEXP * T_TOK * 4); // 8*4096*4
  unsigned short* xb  = (unsigned short*)(ws + 256 + 2ull * NEXP * T_TOK * 4);           // 8 MB
  unsigned short* act = (unsigned short*)(ws + 256 + 2ull * NEXP * T_TOK * 4
                                          + 2ull * T_TOK * HIDDEN);                      // 46 MB

  moe_init_kernel<<<dim3(4096), dim3(256), 0, stream>>>(out, counts);
  moe_router_kernel<<<dim3(T_TOK), dim3(256), 0, stream>>>(hs, lnw, rw, xb,
                                                           counts, token_list, gate_list);
  moe_offsets_kernel<<<dim3(1), dim3(32), 0, stream>>>(counts, offsets);

  dim3 g1(FFN / TN, T_TOK / TM, NEXP);   // 44 x 32 x 8
  moe_ffn1_kernel<<<g1, dim3(256), 0, stream>>>(xb, w1, w3, counts, offsets,
                                                token_list, act);
  dim3 g2(HIDDEN / TN, T_TOK / TM, NEXP); // 16 x 32 x 8
  moe_ffn2_kernel<<<g2, dim3(256), 0, stream>>>(act, w2, counts, offsets,
                                                token_list, gate_list, out);
}
